// Router_85718957294260
// MI455X (gfx1250) — matrix operand order repaired and checked by reading, not compiled, not run on hardware
//
#include <hip/hip_runtime.h>
#include <hip/hip_bf16.h>
#include <math.h>

// ---------------------------------------------------------------------------
// Fused router for MI455X (gfx1250, wave32, WMMA).
//   K1: pack conv weights into per-lane-contiguous WMMA B-fragments (f16),
//       zero the feature accumulator.
//   K2: fused conv1+relu+conv2+relu+spatial-sum as implicit GEMM via
//       v_wmma_f32_16x16x32_f16; all activations stay in LDS.
//   K3: tiny scalar epilogue (MLP / softmax / noisy top-2 gates / load).
// ---------------------------------------------------------------------------

typedef __attribute__((ext_vector_type(16))) _Float16 v16h;
typedef __attribute__((ext_vector_type(8)))  float    v8f;

union F16x16 { v16h v; _Float16 h[16]; };

#define IMG_B 64
#define IMG_H 256
#define IMG_W 256
#define C1    16                 // conv1 out channels
#define C2    32                 // conv2 out channels
#define TS    32                 // output tile (TS x TS)
#define R1    34                 // conv1 output region (TS + 2 halo)
#define RI    36                 // input region (TS + 4 halo)
#define NPIX1 (R1 * R1)          // 1156
#define EXP_E 64
#define DIM_D 32

// ---------------------------------------------------------------------------
// A-fragment builders. HALF (= lane>>4) is a template parameter so every
// k / cin / dy / dx is a compile-time constant: each element lowers to one
// ds_load_u16 with an immediate offset off a single base VGPR.
// Contiguous k order used here (matches the packed B): element e of lane l holds
//   M = l&15,  K = 16*HALF + e.
// ---------------------------------------------------------------------------
template <int HALF>
__device__ __forceinline__ v16h a_conv1(const _Float16* sIn, int aBase) {
  F16x16 A;
#pragma unroll
  for (int e = 0; e < 16; ++e) {
    const int k = HALF * 16 + e;
    if (k < 27) {
      const int cin = k / 9, off = k % 9;
      A.h[e] = sIn[cin * (RI * RI) + aBase + (off / 3) * RI + (off % 3)];
    } else {
      A.h[e] = (_Float16)0.f;     // K padding 27..31
    }
  }
  return A.v;
}

template <int HALF, int KS>
__device__ __forceinline__ v16h a_conv2(const _Float16* sC1, int aBase) {
  F16x16 A;
#pragma unroll
  for (int e = 0; e < 16; ++e) {
    const int k = KS * 32 + HALF * 16 + e;
    if (k < 144) {
      const int cin = k / 9, off = k % 9;
      A.h[e] = sC1[aBase + ((off / 3) * R1 + (off % 3)) * C1 + cin];
    } else {
      A.h[e] = (_Float16)0.f;     // K padding 144..159
    }
  }
  return A.v;
}

// ---------------------------------------------------------------------------
// Kernel 1: pack conv weights into WMMA B-fragment order (lane-contiguous).
// B layout (16-bit, KxN with N=16): lane l = column N (l&15) of half (l>>4);
// element e holds K = 16*(l>>4) + e.  Stored as frag[lane*16 + e].
// ---------------------------------------------------------------------------
__global__ void pack_weights_kernel(const float* __restrict__ c1w,   // (16,3,3,3)
                                    const float* __restrict__ c2w,   // (32,16,3,3)
                                    _Float16* __restrict__ w1f,      // [32][16]
                                    _Float16* __restrict__ w2f,      // [10][32][16]
                                    float* __restrict__ featAcc) {   // (64,32)
  const int tid = threadIdx.x;
  // conv1 fragment (single): K = 27 -> pad 32, N = 16
  for (int i = tid; i < 32 * 16; i += 256) {
    const int l = i >> 4, e = i & 15;
    const int N = l & 15, K = (l >> 4) * 16 + e;
    float v = 0.f;
    if (K < 27) {
      const int cin = K / 9, off = K % 9, ky = off / 3, kx = off % 3;
      v = c1w[((N * 3 + cin) * 3 + ky) * 3 + kx];
    }
    w1f[i] = (_Float16)v;
  }
  // conv2 fragments: f = ks*2 + nt  (5 K-steps x 2 N-tiles), K = 144 -> pad 160
  for (int i = tid; i < 10 * 32 * 16; i += 256) {
    const int e = i & 15, l = (i >> 4) & 31, f = i >> 9;
    const int ks = f >> 1, nt = f & 1;
    const int N = nt * 16 + (l & 15);
    const int K = ks * 32 + (l >> 4) * 16 + e;
    float v = 0.f;
    if (K < 144) {
      const int cin = K / 9, off = K % 9, ky = off / 3, kx = off % 3;
      v = c2w[((N * 16 + cin) * 3 + ky) * 3 + kx];
    }
    w2f[i] = (_Float16)v;
  }
  for (int i = tid; i < IMG_B * C2; i += 256) featAcc[i] = 0.f;
}

// ---------------------------------------------------------------------------
// Kernel 2: fused conv1 -> relu -> conv2 -> relu -> per-channel spatial sum.
// Grid (8,8,64); 256 threads = 8 waves.
// ---------------------------------------------------------------------------
__global__ __launch_bounds__(256) void conv_feat_kernel(
    const float* __restrict__ x,     // (64,3,256,256)
    const float* __restrict__ c1b,   // (16)
    const float* __restrict__ c2b,   // (32)
    const _Float16* __restrict__ w1f,
    const _Float16* __restrict__ w2f,
    float* __restrict__ featAcc) {
  __shared__ __align__(16) _Float16 sIn[3 * RI * RI + 64]; // +slack (garbage lanes suppressed)
  __shared__ __align__(16) _Float16 sC1[NPIX1 * C1];       // 37 KB conv1 activations
  __shared__ __align__(16) _Float16 sW2f[10 * 32 * 16];    // 10 KB conv2 B-fragments
  __shared__ float sSum[C2];

  const int tid  = threadIdx.x;
  const int lane = tid & 31;
  const int wave = tid >> 5;
  const int half = lane >> 4;
  const int ln   = lane & 15;

  const int b  = blockIdx.z;
  const int ox = blockIdx.x * TS;
  const int oy = blockIdx.y * TS;
  const bool edge = (ox == 0) || (oy == 0) || (ox + TS == IMG_W) || (oy + TS == IMG_H);

  // ---- Phase 0: stage conv2 B-fragments (verbatim) + input halo tile ----
  {
    const uint4* src = (const uint4*)w2f;
    uint4*       dst = (uint4*)sW2f;
    for (int i = tid; i < (10 * 32 * 16) / 8; i += 256) dst[i] = src[i];
  }
  if (tid < C2) sSum[tid] = 0.f;
  for (int i = tid; i < 3 * RI * RI; i += 256) {
    const int c = i / (RI * RI);
    const int r = i % (RI * RI);
    const int ry = r / RI, rx = r % RI;
    const int iy = oy - 2 + ry, ix = ox - 2 + rx;
    float v = 0.f;
    if (iy >= 0 && iy < IMG_H && ix >= 0 && ix < IMG_W)
      v = x[((b * 3 + c) * IMG_H + iy) * IMG_W + ix];
    sIn[i] = (_Float16)v;
  }
  // conv1 B-fragment: loop-invariant, 32 B/lane from global, lives in VGPRs
  F16x16 B1;
  B1.v = ((const v16h*)w1f)[lane];
  const float bias1 = c1b[ln];
  __syncthreads();

  // ---- Phase 1: conv1 over 34x34 region (rows x 3 col-tiles, implicit GEMM).
  // Stores are unconditional relu(acc+bias); out-of-image border lines are
  // zeroed in a separate pass below (edge blocks only), keeping this loop lean.
#pragma unroll
  for (int c = 0; c < 3; ++c) {                 // col-tile: compile-time
    for (int r = wave; r < R1; r += 8) {        // region row: wave-uniform
      const int aBase = r * RI + c * 16 + ln;   // A base (halves) in sIn
      const v16h Av = half ? a_conv1<1>(sIn, aBase) : a_conv1<0>(sIn, aBase);
      v8f acc = {};
      acc = __builtin_amdgcn_wmma_f32_16x16x32_f16(false, Av, false, B1.v,
                                                   (short)0, acc, false, false);
      // D element e: M = e + 8*half -> region col = 16c + e + 8*half, N = ln
      const int sBase = (r * R1 + c * 16 + half * 8) * C1 + ln;
#pragma unroll
      for (int e = 0; e < 8; ++e) {
        if (c * 16 + e < R1) {                  // statically skip impossible cols
          const float v = acc[e] + bias1;
          const float rl = v > 0.f ? v : 0.f;
          const int col = c * 16 + e + half * 8;
          if (col < R1)                         // folds true for c<2; c==2 -> half==0
            sC1[sBase + e * C1] = (_Float16)rl;
        }
      }
    }
  }
  __syncthreads();

  // ---- Phase 1b (edge blocks only): zero out-of-image sC1 border lines so
  // conv2's SAME zero-padding is exact. Block-uniform branch. ----
  if (edge) {
    for (int i = tid; i < R1 * C1; i += 256) {
      const int p = i >> 4;          // position along the border line (0..33)
      const int cch = i & 15;        // channel
      if (oy == 0)          sC1[(0 * R1 + p) * C1 + cch]  = (_Float16)0.f;
      if (oy + TS == IMG_H) sC1[(33 * R1 + p) * C1 + cch] = (_Float16)0.f;
      if (ox == 0)          sC1[(p * R1 + 0) * C1 + cch]  = (_Float16)0.f;
      if (ox + TS == IMG_W) sC1[(p * R1 + 33) * C1 + cch] = (_Float16)0.f;
    }
    __syncthreads();
  }

  // ---- Phase 2: conv2 over 32x32 tile (2 N-tiles x 64 M-tiles x 5 K-steps) ----
#define CONV2_STEP(KS)                                                         \
  {                                                                            \
    const v16h Av = half ? a_conv2<1, KS>(sC1, aBase)                          \
                         : a_conv2<0, KS>(sC1, aBase);                         \
    acc = __builtin_amdgcn_wmma_f32_16x16x32_f16(false, Av, false, Bk[KS].v,   \
                                                 (short)0, acc, false, false); \
  }

#pragma unroll
  for (int nt = 0; nt < 2; ++nt) {
    // B-fragments are invariant across the whole M loop: keep in VGPRs.
    F16x16 Bk[5];
#pragma unroll
    for (int ks = 0; ks < 5; ++ks)
      Bk[ks].v = *(const v16h*)&sW2f[((ks * 2 + nt) * 32 + lane) * 16];
    const int ch = nt * 16 + ln;
    const float bias2 = c2b[ch];
    float chSum = 0.f;
    for (int mt = wave; mt < 64; mt += 8) {
      const int p = mt * 16 + ln;               // out-tile pixel (M = ln)
      const int aBase = ((p >> 5) * R1 + (p & 31)) * C1;
      v8f acc = {};
      CONV2_STEP(0)
      CONV2_STEP(1)
      CONV2_STEP(2)
      CONV2_STEP(3)
      CONV2_STEP(4)
#pragma unroll
      for (int e = 0; e < 8; ++e) {
        const float v = acc[e] + bias2;
        chSum += v > 0.f ? v : 0.f;             // relu + spatial sum
      }
    }
    atomicAdd(&sSum[ch], chSum);                // ds_add_f32, 1 per wave x nt
  }
#undef CONV2_STEP

  __syncthreads();
  if (tid < C2) atomicAdd(&featAcc[b * C2 + tid], sSum[tid]);
}

// ---------------------------------------------------------------------------
// Kernel 3: router epilogue. One block, 64 threads (thread = batch row).
// ---------------------------------------------------------------------------
__global__ __launch_bounds__(64) void router_kernel(
    const int* __restrict__ task_id, const float* __restrict__ noise,
    const float* __restrict__ mw1, const float* __restrict__ mb1,
    const float* __restrict__ mw2, const float* __restrict__ mb2,
    const float* __restrict__ degw, const float* __restrict__ degb,
    const float* __restrict__ ek,   const float* __restrict__ gw,
    const float* __restrict__ gb,   const float* __restrict__ nw,
    const float* __restrict__ nb,   const float* __restrict__ featAcc,
    float* __restrict__ out) {      // gates (64*64) then load (64)
  const int b = threadIdx.x;
  float feat[DIM_D], comb[DIM_D], h1[DIM_D];
  const float invHW = 1.f / (float)(IMG_H * IMG_W);
  for (int c = 0; c < DIM_D; ++c) feat[c] = featAcc[b * DIM_D + c] * invHW;

  const float t = (float)task_id[b];
  for (int j = 0; j < DIM_D; ++j) {
    const float v = t * mw1[j] + mb1[j];
    h1[j] = v > 0.f ? v : 0.f;
  }
  for (int i = 0; i < DIM_D; ++i) {
    float te = mb2[i], de = degb[i];
    for (int j = 0; j < DIM_D; ++j) te += h1[j] * mw2[j * DIM_D + i];
    for (int c = 0; c < DIM_D; ++c) de += feat[c] * degw[c * DIM_D + i];
    comb[i] = te + 0.2f * de;                  // (1 - ALPHA)
  }

  float ew[EXP_E];
  float mx = -1e30f;
  const float scale = 0.1767766952966369f;     // 1/sqrt(32)
  for (int e = 0; e < EXP_E; ++e) {
    float s = 0.f;
    for (int i = 0; i < DIM_D; ++i) s += comb[i] * ek[e * DIM_D + i];
    s *= scale;
    ew[e] = s;
    mx = fmaxf(mx, s);
  }
  float den = 0.f;
  for (int e = 0; e < EXP_E; ++e) { ew[e] = __expf(ew[e] - mx); den += ew[e]; }
  const float inv = 1.f / den;
  for (int e = 0; e < EXP_E; ++e) ew[e] *= inv;

  float logit[EXP_E];
  for (int e = 0; e < EXP_E; ++e) {
    float cl = gb[e], st = nb[e];
    for (int k = 0; k < EXP_E; ++k) {
      cl += ew[k] * gw[k * EXP_E + e];
      st += ew[k] * nw[k * EXP_E + e];
    }
    const float sp = (st > 20.f) ? st : log1pf(expf(st));  // softplus
    logit[e] = cl + noise[b * EXP_E + e] * (sp + 0.01f);
  }

  // top-2 (strict > keeps lowest index on ties, matching lax.top_k)
  int i1 = 0; float l1 = logit[0];
  for (int e = 1; e < EXP_E; ++e) if (logit[e] > l1) { l1 = logit[e]; i1 = e; }
  int i2 = (i1 == 0) ? 1 : 0; float l2 = logit[i2];
  for (int e = 0; e < EXP_E; ++e)
    if (e != i1 && logit[e] > l2) { l2 = logit[e]; i2 = e; }

  const float m2 = fmaxf(l1, l2);
  const float e1 = __expf(l1 - m2), e2 = __expf(l2 - m2);
  const float g1 = e1 / (e1 + e2), g2 = e2 / (e1 + e2);

  for (int e = 0; e < EXP_E; ++e) out[b * EXP_E + e] = 0.f;
  out[b * EXP_E + i1] = g1;
  out[b * EXP_E + i2] = g2;

  out[IMG_B * EXP_E + b] = 0.f;                // zero load[] (single block)
  __syncthreads();
  atomicAdd(&out[IMG_B * EXP_E + i1], g1);
  atomicAdd(&out[IMG_B * EXP_E + i2], g2);
}

// ---------------------------------------------------------------------------
extern "C" void kernel_launch(void* const* d_in, const int* in_sizes, int n_in,
                              void* d_out, int out_size, void* d_ws, size_t ws_size,
                              hipStream_t stream) {
  const int*   task_id  = (const int*)  d_in[0];
  const float* deg      = (const float*)d_in[1];
  const float* noise    = (const float*)d_in[2];
  const float* mlp_w1   = (const float*)d_in[3];
  const float* mlp_b1   = (const float*)d_in[4];
  const float* mlp_w2   = (const float*)d_in[5];
  const float* mlp_b2   = (const float*)d_in[6];
  const float* conv1_w  = (const float*)d_in[7];
  const float* conv1_b  = (const float*)d_in[8];
  const float* conv2_w  = (const float*)d_in[9];
  const float* conv2_b  = (const float*)d_in[10];
  const float* deg_w    = (const float*)d_in[11];
  const float* deg_b    = (const float*)d_in[12];
  const float* ekeys    = (const float*)d_in[13];
  const float* gate_w   = (const float*)d_in[14];
  const float* gate_b   = (const float*)d_in[15];
  const float* noise_w  = (const float*)d_in[16];
  const float* noise_b  = (const float*)d_in[17];

  char* ws = (char*)d_ws;
  _Float16* w1f     = (_Float16*)(ws);                  // 32*16  f16 =  1 KB
  _Float16* w2f     = (_Float16*)(ws + 1024);           // 10*32*16 f16 = 10 KB
  float*    featAcc = (float*)   (ws + 1024 + 10240);   // 64*32 f32 = 8 KB

  pack_weights_kernel<<<1, 256, 0, stream>>>(conv1_w, conv2_w, w1f, w2f, featAcc);
  conv_feat_kernel<<<dim3(IMG_W / TS, IMG_H / TS, IMG_B), 256, 0, stream>>>(
      deg, conv1_b, conv2_b, w1f, w2f, featAcc);
  router_kernel<<<1, 64, 0, stream>>>(task_id, noise, mlp_w1, mlp_b1, mlp_w2,
                                      mlp_b2, deg_w, deg_b, ekeys, gate_w,
                                      gate_b, noise_w, noise_b, featAcc,
                                      (float*)d_out);
}
